// Mamba_45758581571908
// MI455X (gfx1250) — compile-verified, matrix operand order checked by reading, not run on hardware
//
#include <hip/hip_runtime.h>

#define L_SEQ 2048
#define DM    512
#define DI    1024
#define DS    16
#define DTR   32

typedef __attribute__((ext_vector_type(16))) __bf16       v16bf;
typedef __attribute__((ext_vector_type(8)))  float        v8f;
typedef __attribute__((ext_vector_type(4)))  float        f4;
typedef __attribute__((ext_vector_type(4)))  unsigned int u4;

#if __has_builtin(__builtin_amdgcn_global_load_async_to_lds_b128) && \
    __has_builtin(__builtin_amdgcn_s_wait_asynccnt)
#define HAS_ASYNC_LDS 1
#else
#define HAS_ASYNC_LDS 0
#endif

#if HAS_ASYNC_LDS
typedef int i4g __attribute__((vector_size(16)));             // v4i32
typedef __attribute__((address_space(1))) i4g* gp_b128;       // global v4i32*
typedef __attribute__((address_space(3))) i4g* lp_b128;       // LDS    v4i32*
#endif

// fp32 -> bf16, round-to-nearest-even
static __device__ __forceinline__ unsigned short f2bf(float x) {
  union { float f; unsigned int u; } v; v.f = x;
  unsigned int r = v.u + 0x7fffu + ((v.u >> 16) & 1u);
  return (unsigned short)(r >> 16);
}
static __device__ __forceinline__ unsigned int pack2(float a, float b) {
  return (unsigned int)f2bf(a) | ((unsigned int)f2bf(b) << 16);
}

// 16-byte global -> LDS copy.  On gfx1250 toolchains with the async builtins
// this emits GLOBAL_LOAD_ASYNC_TO_LDS_B128 (ASYNCcnt-tracked DMA, no VGPR
// round trip); otherwise a plain load+ds_store fallback.
static __device__ __forceinline__ void cp_g2l_16(unsigned short* ldst,
                                                 const unsigned short* gsrc) {
#if HAS_ASYNC_LDS
  __builtin_amdgcn_global_load_async_to_lds_b128(
      (gp_b128)gsrc, (lp_b128)ldst, 0, 0);
#else
  *reinterpret_cast<u4*>(ldst) = *reinterpret_cast<const u4*>(gsrc);
#endif
}
static __device__ __forceinline__ void wait_g2l() {
#if HAS_ASYNC_LDS
  __builtin_amdgcn_s_wait_asynccnt(0);
#endif
}

// Assemble one WMMA A/B fragment (16x32 / 32x16, bf16) from an LDS tile with
// K contiguous per row and row stride `rs` halfwords. Per ISA 7.12.2: lanes
// 0-15 hold K=[0..7],[16..23]; lanes 16-31 hold K=[8..15],[24..31].
static __device__ __forceinline__ v16bf frag_ld(const unsigned short* lds,
                                                int row, int rs, int lane) {
  const int m  = lane & 15;
  const int k0 = (lane & 16) ? 8 : 0;
  union { v16bf v; u4 u[2]; } f;
  const unsigned short* p = lds + (row + m) * rs;
  f.u[0] = *reinterpret_cast<const u4*>(p + k0);
  f.u[1] = *reinterpret_cast<const u4*>(p + k0 + 16);
  return f.v;
}

static __device__ __forceinline__ v8f wmma_bf16(v16bf a, v16bf b, v8f c) {
  return __builtin_amdgcn_wmma_f32_16x16x32_bf16(false, a, false, b,
                                                 (short)0, c, false, false);
}

// ---------------------------------------------------------------------------
// P0: fp32 -> bf16 pack (one shot; removes all conversion VALU from GEMM hot
// loops and halves GEMM-side HBM read bytes).
// ---------------------------------------------------------------------------
__global__ __launch_bounds__(256) void k_pack_bf16(const float* __restrict__ src,
                                                   unsigned short* __restrict__ dst,
                                                   int n4) {
  int i = blockIdx.x * 256 + threadIdx.x;
  if (i < n4) {
    f4 v = *reinterpret_cast<const f4*>(src + (size_t)i * 4);
    unsigned int* d = reinterpret_cast<unsigned int*>(dst) + (size_t)i * 2;
    d[0] = pack2(v.x, v.y);
    d[1] = pack2(v.z, v.w);
  }
}

// ---------------------------------------------------------------------------
// K1: xz2[s][ch][l] = sum_d hs[b][l][d] * w_in[e][d] (bf16 operands), with the
// e->(s,ch) stream split and L-reversal for reverse streams folded into the
// store.  Async double-buffered global->LDS: copy of K-tile k+1 overlaps the
// WMMA clause of tile k; stage sync = s_wait_asynccnt 0 + barrier.
// ---------------------------------------------------------------------------
__global__ __launch_bounds__(256) void k_gemm_xz(const unsigned short* __restrict__ hsb16,
                                                 const unsigned short* __restrict__ wbf,
                                                 float* __restrict__ xz2) {
  __shared__ unsigned short lA[2][128 * 32];
  __shared__ unsigned short lB[2][64 * 32];
  const int tid = threadIdx.x, wave = tid >> 5, lane = tid & 31;
  const int l0 = blockIdx.x * 64;
  const int e0 = blockIdx.y * 128;
  const int b  = blockIdx.z;
  const unsigned short* hb = hsb16 + (size_t)b * L_SEQ * DM;
  v8f acc[4] = {};

  auto issue = [&](int kk, int buf) {
    const int k0 = kk * 32;
#pragma unroll
    for (int i = 0; i < 2; ++i) {              // A tile: 128x32 bf16 = 8KB
      int f = tid + i * 256;
      int row = f >> 2, c = (f & 3) << 3;
      cp_g2l_16(&lA[buf][row * 32 + c], wbf + (size_t)(e0 + row) * DM + k0 + c);
    }
    {                                          // B tile: 64x32 bf16 = 4KB
      int row = tid >> 2, c = (tid & 3) << 3;
      cp_g2l_16(&lB[buf][row * 32 + c], hb + (size_t)(l0 + row) * DM + k0 + c);
    }
  };

  issue(0, 0);
  for (int kk = 0; kk < DM / 32; ++kk) {
    wait_g2l();
    __syncthreads();                            // stage kk resident in LDS
    if (kk + 1 < DM / 32) issue(kk + 1, (kk + 1) & 1);  // overlap next copy
    const unsigned short* bA = lA[kk & 1];
    const unsigned short* bB = lB[kk & 1];
    v16bf a = frag_ld(bA, wave * 16, 32, lane);
#pragma unroll
    for (int j = 0; j < 4; ++j) {
      v16bf bf = frag_ld(bB, j * 16, 32, lane);
      acc[j] = wmma_bf16(a, bf, acc[j]);
    }
  }
  const int mlo = (lane & 16) ? 8 : 0;
#pragma unroll
  for (int j = 0; j < 4; ++j) {
    const int l = l0 + j * 16 + (lane & 15);
#pragma unroll
    for (int r = 0; r < 8; ++r) {
      int e = e0 + wave * 16 + r + mlo;
      int s, ch, lo;
      if (e < 2 * DI) { s = b;     ch = e;          lo = l; }
      else            { s = 2 + b; ch = e - 2 * DI; lo = L_SEQ - 1 - l; }
      xz2[((size_t)s * 2 * DI + ch) * L_SEQ + lo] = acc[j][r];
    }
  }
}

// ---------------------------------------------------------------------------
// K2: depthwise causal conv1d (d_conv=4) + SiLU; rev=1 runs on L-reversed input.
// ---------------------------------------------------------------------------
__global__ __launch_bounds__(256) void k_conv_silu(const float* __restrict__ xz2,
                                                   const float* __restrict__ cw,
                                                   const float* __restrict__ cb,
                                                   float* __restrict__ xo, int rev) {
  const int l = blockIdx.x * 256 + threadIdx.x;
  const int d = blockIdx.y;
  const int s = blockIdx.z;
  const float* xin = xz2 + ((size_t)s * 2 * DI + d) * L_SEQ;
  const float wj[4] = {cw[d * 4 + 0], cw[d * 4 + 1], cw[d * 4 + 2], cw[d * 4 + 3]};
  float acc = cb[d];
#pragma unroll
  for (int j = 0; j < 4; ++j) {
    int t = l - 3 + j;
    float xv = 0.f;
    if (t >= 0) { int ti = rev ? (L_SEQ - 1 - t) : t; xv = xin[ti]; }
    acc += wj[j] * xv;
  }
  float y = acc / (1.f + __expf(-acc));  // silu
  xo[((size_t)s * DI + d) * L_SEQ + l] = y;
}

// ---------------------------------------------------------------------------
// K3: x_dbl = xproj_w (64x1024, bf16 prepacked, async-staged) * x (1024 x L).
// B transposed in LDS (rows of x are L-contiguous); rs=40 halves keeps 16B
// alignment and dodges bank conflicts.
// ---------------------------------------------------------------------------
__global__ __launch_bounds__(256) void k_gemm_xdbl(const unsigned short* __restrict__ xwbf,
                                                   const float* __restrict__ x,
                                                   float* __restrict__ xdbl) {
  __shared__ unsigned short lA[64 * 32];
  __shared__ unsigned short lBt[128 * 40];
  const int tid = threadIdx.x, wave = tid >> 5, lane = tid & 31;
  const int l0 = blockIdx.x * 128;
  const int s  = blockIdx.y;
  const int mt  = wave & 3;
  const int nt0 = (wave >> 2) * 4;
  const float* xs = x + (size_t)s * DI * L_SEQ;
  v8f acc[4] = {};
  for (int kk = 0; kk < DI / 32; ++kk) {
    const int k0 = kk * 32;
    {                                           // A: 64x32 bf16, async DMA
      int row = tid >> 2, c = (tid & 3) << 3;
      cp_g2l_16(&lA[row * 32 + c], xwbf + (size_t)row * DI + k0 + c);
    }
#pragma unroll
    for (int i = 0; i < 4; ++i) {               // Bt[n][k] = x[k0+k][l0+n]
      int f = tid + i * 256;
      int k = f >> 5, n0 = (f & 31) << 2;
      f4 v = *reinterpret_cast<const f4*>(xs + (size_t)(k0 + k) * L_SEQ + l0 + n0);
      lBt[(n0 + 0) * 40 + k] = f2bf(v.x);
      lBt[(n0 + 1) * 40 + k] = f2bf(v.y);
      lBt[(n0 + 2) * 40 + k] = f2bf(v.z);
      lBt[(n0 + 3) * 40 + k] = f2bf(v.w);
    }
    wait_g2l();
    __syncthreads();
    v16bf a = frag_ld(lA, mt * 16, 32, lane);
#pragma unroll
    for (int j = 0; j < 4; ++j) {
      v16bf bf = frag_ld(lBt, (nt0 + j) * 16, 40, lane);
      acc[j] = wmma_bf16(a, bf, acc[j]);
    }
    __syncthreads();
  }
  const int mlo = (lane & 16) ? 8 : 0;
#pragma unroll
  for (int j = 0; j < 4; ++j) {
    int l = l0 + (nt0 + j) * 16 + (lane & 15);
#pragma unroll
    for (int r = 0; r < 8; ++r) {
      int m = mt * 16 + r + mlo;
      xdbl[((size_t)s * (DTR + 2 * DS) + m) * L_SEQ + l] = acc[j][r];
    }
  }
}

// ---------------------------------------------------------------------------
// K4: delta = softplus(dtproj_w (1024x32, bf16 async-staged) * dt_r (32 x L)
// + db).  K=32 is a single WMMA step; softplus fused into the epilogue.
// ---------------------------------------------------------------------------
__global__ __launch_bounds__(256) void k_gemm_delta(const unsigned short* __restrict__ dwbf,
                                                    const float* __restrict__ xdbl,
                                                    const float* __restrict__ db,
                                                    float* __restrict__ delta) {
  __shared__ unsigned short lA[128 * 32];
  __shared__ unsigned short lBt[64 * 40];
  const int tid = threadIdx.x, wave = tid >> 5, lane = tid & 31;
  const int l0 = blockIdx.x * 64;
  const int m0 = blockIdx.y * 128;
  const int s  = blockIdx.z;
  v8f acc[4] = {};
#pragma unroll
  for (int i = 0; i < 2; ++i) {                 // A: 128x32 bf16, async DMA
    int f = tid + i * 256;
    int row = f >> 2, c = (f & 3) << 3;
    cp_g2l_16(&lA[row * 32 + c], dwbf + (size_t)(m0 + row) * DTR + c);
  }
#pragma unroll
  for (int i = 0; i < 2; ++i) {                 // Bt[n][k] = dt_r[k][l0+n]
    int f = tid + i * 256;
    int k = f >> 4, n0 = (f & 15) << 2;
    f4 v = *reinterpret_cast<const f4*>(xdbl + ((size_t)s * 64 + k) * L_SEQ + l0 + n0);
    lBt[(n0 + 0) * 40 + k] = f2bf(v.x);
    lBt[(n0 + 1) * 40 + k] = f2bf(v.y);
    lBt[(n0 + 2) * 40 + k] = f2bf(v.z);
    lBt[(n0 + 3) * 40 + k] = f2bf(v.w);
  }
  wait_g2l();
  __syncthreads();
  v16bf a = frag_ld(lA, wave * 16, 32, lane);
#pragma unroll
  for (int j = 0; j < 4; ++j) {
    v16bf bf = frag_ld(lBt, j * 16, 40, lane);
    acc[j] = wmma_bf16(a, bf, acc[j]);
  }
  const int mlo = (lane & 16) ? 8 : 0;
#pragma unroll
  for (int j = 0; j < 4; ++j) {
    int l = l0 + j * 16 + (lane & 15);
#pragma unroll
    for (int r = 0; r < 8; ++r) {
      int d = m0 + wave * 16 + r + mlo;
      float v = acc[j][r] + db[d];
      float sp = (v > 15.f) ? v : __logf(1.f + __expf(v));
      delta[((size_t)s * DI + d) * L_SEQ + l] = sp;
    }
  }
}

// ---------------------------------------------------------------------------
// K5: selective scan + D-skip + z-gate.  One thread per (stream, channel),
// 16-state in registers, float4 row loads of delta/x, wave-uniform broadcast
// loads of B/C.  Output position handles all bidirectional reversals; the
// backward pass accumulates (+=) into the combined buffer.
// ---------------------------------------------------------------------------
__global__ __launch_bounds__(256) void k_scan(const float* __restrict__ delta,
                                              const float* __restrict__ x,
                                              const float* __restrict__ xdbl,
                                              const float* __restrict__ xz2,
                                              const float* __restrict__ Alog,
                                              const float* __restrict__ Dvec,
                                              float* __restrict__ out,
                                              int bwd_call, int maskA) {
  const int idx = blockIdx.x * 256 + threadIdx.x;
  const int s = idx >> 10;
  const int d = idx & (DI - 1);
  float Aa[DS];
#pragma unroll
  for (int n = 0; n < DS; ++n) {
    float a = -__expf(Alog[d * DS + n]);
    if (maskA && d >= DI / 2 && d < DI / 2 + DS && n == (d - DI / 2)) a = 0.f;
    Aa[n] = a;
  }
  const float Dv = Dvec[d];
  const float* pd = delta + ((size_t)s * DI + d) * L_SEQ;
  const float* px = x     + ((size_t)s * DI + d) * L_SEQ;
  const float* pB = xdbl  + ((size_t)s * 64 + DTR) * L_SEQ;
  const float* pC = xdbl  + ((size_t)s * 64 + DTR + DS) * L_SEQ;
  const float* pz = xz2   + ((size_t)s * 2 * DI + DI + d) * L_SEQ;
  const int batch = s & 1;
  const int ch = (s < 2) ? d : (DI + d);
  float* po = out + ((size_t)batch * 2 * DI + ch) * L_SEQ;
  float h[DS];
#pragma unroll
  for (int n = 0; n < DS; ++n) h[n] = 0.f;

  for (int l4 = 0; l4 < L_SEQ; l4 += 4) {
    f4 dv4 = *reinterpret_cast<const f4*>(pd + l4);
    f4 xv4 = *reinterpret_cast<const f4*>(px + l4);
#pragma unroll
    for (int q = 0; q < 4; ++q) {
      const int l = l4 + q;
      const float dv = dv4[q], xv = xv4[q];
      const float dx = dv * xv;
      float y = 0.f;
#pragma unroll
      for (int n = 0; n < DS; ++n) {
        h[n] = h[n] * __expf(dv * Aa[n]) + dx * pB[(size_t)n * L_SEQ + l];
        y += h[n] * pC[(size_t)n * L_SEQ + l];
      }
      const int lz = bwd_call ? (L_SEQ - 1 - l) : l;
      const float zv = pz[lz];
      const float g = zv / (1.f + __expf(-zv));
      const float yy = (y + Dv * xv) * g;
      int lo;
      if (s < 2) lo = bwd_call ? (L_SEQ - 1 - l) : l;
      else       lo = bwd_call ? l : (L_SEQ - 1 - l);
      if (bwd_call) po[lo] += yy; else po[lo] = yy;
    }
  }
}

// ---------------------------------------------------------------------------
// K6: result[b][l][o] = sum_d out[b][d][l] * w_out[o][d].  A = w_out (bf16
// prepacked, async-staged), B transposed in LDS.
// ---------------------------------------------------------------------------
__global__ __launch_bounds__(256) void k_gemm_out(const unsigned short* __restrict__ wobf,
                                                  const float* __restrict__ oc,
                                                  float* __restrict__ res) {
  __shared__ unsigned short lA[128 * 32];
  __shared__ unsigned short lBt[64 * 40];
  const int tid = threadIdx.x, wave = tid >> 5, lane = tid & 31;
  const int l0 = blockIdx.x * 64;
  const int m0 = blockIdx.y * 128;
  const int b  = blockIdx.z;
  const float* ocb = oc + (size_t)b * 2 * DI * L_SEQ;
  v8f acc[4] = {};
  for (int kk = 0; kk < (2 * DI) / 32; ++kk) {
    const int k0 = kk * 32;
#pragma unroll
    for (int i = 0; i < 2; ++i) {               // A: 128x32 bf16, async DMA
      int f = tid + i * 256;
      int row = f >> 2, c = (f & 3) << 3;
      cp_g2l_16(&lA[row * 32 + c], wobf + (size_t)(m0 + row) * (2 * DI) + k0 + c);
    }
#pragma unroll
    for (int i = 0; i < 2; ++i) {               // Bt[n][k] = oc[b][k0+k][l0+n]
      int f = tid + i * 256;
      int k = f >> 4, n0 = (f & 15) << 2;
      f4 v = *reinterpret_cast<const f4*>(ocb + (size_t)(k0 + k) * L_SEQ + l0 + n0);
      lBt[(n0 + 0) * 40 + k] = f2bf(v.x);
      lBt[(n0 + 1) * 40 + k] = f2bf(v.y);
      lBt[(n0 + 2) * 40 + k] = f2bf(v.z);
      lBt[(n0 + 3) * 40 + k] = f2bf(v.w);
    }
    wait_g2l();
    __syncthreads();
    v16bf a = frag_ld(lA, wave * 16, 32, lane);
#pragma unroll
    for (int j = 0; j < 4; ++j) {
      v16bf bf = frag_ld(lBt, j * 16, 40, lane);
      acc[j] = wmma_bf16(a, bf, acc[j]);
    }
    __syncthreads();
  }
  const int mlo = (lane & 16) ? 8 : 0;
#pragma unroll
  for (int j = 0; j < 4; ++j) {
    int l = l0 + j * 16 + (lane & 15);
#pragma unroll
    for (int r = 0; r < 8; ++r) {
      int o = m0 + wave * 16 + r + mlo;
      res[((size_t)b * L_SEQ + l) * DM + o] = acc[j][r];
    }
  }
}

extern "C" void kernel_launch(void* const* d_in, const int* in_sizes, int n_in,
                              void* d_out, int out_size, void* d_ws, size_t ws_size,
                              hipStream_t stream) {
  const float* hs    = (const float*)d_in[0];
  const float* w_in  = (const float*)d_in[1];
  const float* cw_f  = (const float*)d_in[2];
  const float* cb_f  = (const float*)d_in[3];
  const float* xw_f  = (const float*)d_in[4];
  const float* dw_f  = (const float*)d_in[5];
  const float* db_f  = (const float*)d_in[6];
  const float* Alog  = (const float*)d_in[7];
  const float* D_f   = (const float*)d_in[8];
  const float* cw_b  = (const float*)d_in[9];
  const float* cb_b  = (const float*)d_in[10];
  const float* xw_b  = (const float*)d_in[11];
  const float* dw_b  = (const float*)d_in[12];
  const float* db_b  = (const float*)d_in[13];
  const float* Ablog = (const float*)d_in[14];
  const float* D_b   = (const float*)d_in[15];
  const float* w_out = (const float*)d_in[16];

  float* ws   = (float*)d_ws;
  float* xz2  = ws;                                  // (4, 2048, 2048)  64 MB
  float* xbuf = xz2  + (size_t)4 * 2 * DI * L_SEQ;   // (4, 1024, 2048)  32 MB
  float* xdbl = xbuf + (size_t)4 * DI * L_SEQ;       // (4,   64, 2048)   2 MB
  float* delt = xdbl + (size_t)4 * 64 * L_SEQ;       // (4, 1024, 2048)  32 MB
  float* outc = delt + (size_t)4 * DI * L_SEQ;       // (2, 2048, 2048)  32 MB
  unsigned short* bfarea = (unsigned short*)(outc + (size_t)2 * 2 * DI * L_SEQ);
  unsigned short* hs_bf   = bfarea;                          // 2M halves
  unsigned short* win_bf  = hs_bf  + (size_t)2 * L_SEQ * DM; // 2M halves
  unsigned short* wout_bf = win_bf + (size_t)4 * DI * DM;    // 1M halves
  unsigned short* xwf_bf  = wout_bf + (size_t)DM * 2 * DI;   // 64K
  unsigned short* xwb_bf  = xwf_bf + (size_t)64 * DI;
  unsigned short* dwf_bf  = xwb_bf + (size_t)64 * DI;        // 32K
  unsigned short* dwb_bf  = dwf_bf + (size_t)DI * DTR;

  // bf16 prepack (one shot)
  k_pack_bf16<<<dim3((2 * L_SEQ * DM) / 4 / 256), 256, 0, stream>>>(hs, hs_bf, (2 * L_SEQ * DM) / 4);
  k_pack_bf16<<<dim3((4 * DI * DM) / 4 / 256), 256, 0, stream>>>(w_in, win_bf, (4 * DI * DM) / 4);
  k_pack_bf16<<<dim3((DM * 2 * DI) / 4 / 256), 256, 0, stream>>>(w_out, wout_bf, (DM * 2 * DI) / 4);
  k_pack_bf16<<<dim3((64 * DI) / 4 / 256), 256, 0, stream>>>(xw_f, xwf_bf, (64 * DI) / 4);
  k_pack_bf16<<<dim3((64 * DI) / 4 / 256), 256, 0, stream>>>(xw_b, xwb_bf, (64 * DI) / 4);
  k_pack_bf16<<<dim3((DI * DTR) / 4 / 256), 256, 0, stream>>>(dw_f, dwf_bf, (DI * DTR) / 4);
  k_pack_bf16<<<dim3((DI * DTR) / 4 / 256), 256, 0, stream>>>(dw_b, dwb_bf, (DI * DTR) / 4);

  // input projection into stream layout (with reversal for streams 2,3)
  k_gemm_xz<<<dim3(L_SEQ / 64, (4 * DI) / 128, 2), 256, 0, stream>>>(hs_bf, win_bf, xz2);

  // ---- forward direction ----
  k_conv_silu <<<dim3(L_SEQ / 256, DI, 4), 256, 0, stream>>>(xz2, cw_f, cb_f, xbuf, 0);
  k_gemm_xdbl <<<dim3(L_SEQ / 128, 4),     256, 0, stream>>>(xwf_bf, xbuf, xdbl);
  k_gemm_delta<<<dim3(L_SEQ / 64, DI / 128, 4), 256, 0, stream>>>(dwf_bf, xdbl, db_f, delt);
  k_scan      <<<dim3((4 * DI) / 256),     256, 0, stream>>>(delt, xbuf, xdbl, xz2,
                                                             Alog, D_f, outc, 0, 1);
  // ---- backward direction (buffers reused) ----
  k_conv_silu <<<dim3(L_SEQ / 256, DI, 4), 256, 0, stream>>>(xz2, cw_b, cb_b, xbuf, 1);
  k_gemm_xdbl <<<dim3(L_SEQ / 128, 4),     256, 0, stream>>>(xwb_bf, xbuf, xdbl);
  k_gemm_delta<<<dim3(L_SEQ / 64, DI / 128, 4), 256, 0, stream>>>(dwb_bf, xdbl, db_b, delt);
  k_scan      <<<dim3((4 * DI) / 256),     256, 0, stream>>>(delt, xbuf, xdbl, xz2,
                                                             Ablog, D_b, outc, 1, 0);
  // output projection
  k_gemm_out<<<dim3(L_SEQ / 64, DM / 128, 2), 256, 0, stream>>>(wout_bf, outc, (float*)d_out);

  (void)in_sizes; (void)n_in; (void)out_size; (void)ws_size;
}
